// SAGE_67053029425277
// MI455X (gfx1250) — compile-verified, matrix operand order checked by reading, not run on hardware
//
#include <hip/hip_runtime.h>
#include <hip/hip_bf16.h>

typedef __attribute__((ext_vector_type(16))) __bf16 v16bf;
typedef __attribute__((ext_vector_type(8)))  __bf16 v8bf;
typedef __attribute__((ext_vector_type(8)))  float  v8f;
typedef __attribute__((ext_vector_type(4)))  float  v4f;

#define NFEAT  128
#define NHID   128
#define NCLASS 64

// ---------------------------------------------------------------------------
// Fragment loaders.
// 16-bit A-matrix 16x32 layout (ISA 7.12.2): lane L holds row M = L%16.
// Lanes 0-15 hold K = {kbase+0..7, kbase+16..23}; lanes 16-31 hold
// K = {kbase+8..15, kbase+24..31}.  Caller passes p already offset by
// (lane>>4)*8, so we read 8 consecutive + 8 consecutive at +16.
// ---------------------------------------------------------------------------
__device__ __forceinline__ v16bf frag_from_f32(const float* __restrict__ p) {
  v4f a0 = *(const v4f*)(p);
  v4f a1 = *(const v4f*)(p + 4);
  v4f a2 = *(const v4f*)(p + 16);
  v4f a3 = *(const v4f*)(p + 20);
  v16bf r;
#pragma unroll
  for (int i = 0; i < 4; ++i) {
    r[i]      = (__bf16)a0[i];
    r[i + 4]  = (__bf16)a1[i];
    r[i + 8]  = (__bf16)a2[i];
    r[i + 12] = (__bf16)a3[i];
  }
  return r;
}

__device__ __forceinline__ v16bf frag_from_lds(const __bf16* p) {
  v8bf b0 = *(const v8bf*)(p);        // ds_load_b128
  v8bf b1 = *(const v8bf*)(p + 16);   // ds_load_b128
  v16bf r;
#pragma unroll
  for (int i = 0; i < 8; ++i) { r[i] = b0[i]; r[i + 8] = b1[i]; }
  return r;
}

// ---------------------------------------------------------------------------
// Zero-fill (agg accumulator must start at 0 for the atomic SpMM).
// ---------------------------------------------------------------------------
__global__ __launch_bounds__(256) void sage_zero(float* __restrict__ p, long n4) {
  long i = ((long)blockIdx.x * blockDim.x + threadIdx.x) * 4;
  if (i < n4) *(v4f*)(p + i) = (v4f){0.f, 0.f, 0.f, 0.f};
}

// ---------------------------------------------------------------------------
// SpMM: agg[r] += vals[e] * feat[cols[e]]  (rows sorted; register-accumulate
// per wave chunk, flush with hardware f32 atomics only on row change).
// Wave = 128 edges, lane = 4 feature columns.  Edge metadata is loaded
// vectorized (one edge per lane per 32-edge chunk) and broadcast with
// v_readlane via __shfl -- removes the serialized per-edge scalar loads.
// ---------------------------------------------------------------------------
__global__ __launch_bounds__(256) void sage_spmm(
    const int* __restrict__ rows, const int* __restrict__ cols,
    const float* __restrict__ vals, const float* __restrict__ feat,
    float* __restrict__ agg, int nedges)
{
  const int lane = threadIdx.x & 31;
  const long wave = (long)blockIdx.x * 8 + (threadIdx.x >> 5);
  const long e0 = wave * 128;
  if (e0 >= nedges) return;
  const long e1 = (e0 + 128 < (long)nedges) ? (e0 + 128) : (long)nedges;
  const int f = lane * 4;

  v4f acc = {0.f, 0.f, 0.f, 0.f};
  int cur = rows[e0];

  for (long ebase = e0; ebase < e1; ebase += 32) {
    const int n = (int)(((e1 - ebase) < 32) ? (e1 - ebase) : 32);
    const long li = ebase + ((lane < n) ? lane : (n - 1));
    const int   rv = rows[li];
    const int   cv = cols[li];
    const float vv = vals[li];
    if (ebase + 32 < e1) {  // prefetch next metadata chunk
      __builtin_prefetch(&rows[ebase + 32], 0, 0);
      __builtin_prefetch(&cols[ebase + 32], 0, 0);
      __builtin_prefetch(&vals[ebase + 32], 0, 0);
    }
#pragma unroll 8
    for (int i = 0; i < n; ++i) {
      const int   r = __shfl(rv, i, 32);   // wave-uniform
      const int   c = __shfl(cv, i, 32);
      const float v = __shfl(vv, i, 32);
      if (r != cur) {                      // uniform (scalar) branch
        float* dst = agg + (size_t)cur * NHID + f;
#pragma unroll
        for (int j = 0; j < 4; ++j)
          __hip_atomic_fetch_add(dst + j, acc[j], __ATOMIC_RELAXED,
                                 __HIP_MEMORY_SCOPE_AGENT);
        acc = (v4f){0.f, 0.f, 0.f, 0.f};
        cur = r;
      }
      const v4f xv = *(const v4f*)(feat + (size_t)c * NHID + f);  // L2-hit gather
#pragma unroll
      for (int j = 0; j < 4; ++j) acc[j] += v * xv[j];
    }
  }
  float* dst = agg + (size_t)cur * NHID + f;
#pragma unroll
  for (int j = 0; j < 4; ++j)
    __hip_atomic_fetch_add(dst + j, acc[j], __ATOMIC_RELAXED,
                           __HIP_MEMORY_SCOPE_AGENT);
}

// ---------------------------------------------------------------------------
// Layer 1: h = relu(l1norm([agg|x](32x256) @ [Wl^T;Wr^T](256x128) + b))
// One wave per PAIR of 16-row tiles (32 nodes): each B fragment pulled from
// LDS feeds TWO independent v_wmma ops (halves LDS traffic per WMMA, and the
// two WMMAs pipeline without D->A/B hazards).  8 WMMA pairs (one per N-tile) per k-step.
// ---------------------------------------------------------------------------
__global__ __launch_bounds__(256) void sage_gemm1(
    const float* __restrict__ agg, const float* __restrict__ x,
    const float* __restrict__ Wl, const float* __restrict__ bl,
    const float* __restrict__ Wr, const float* __restrict__ br,
    float* __restrict__ h, int nnodes)
{
  __shared__ __align__(16) __bf16 lw[NHID * 256];  // [n][k], 64 KB
  const int tid = threadIdx.x;
#pragma unroll 4
  for (int i = tid; i < NHID * 256; i += 256) {
    const int n = i >> 8, k = i & 255;
    const float w = (k < NFEAT) ? Wl[n * NFEAT + k] : Wr[n * NFEAT + (k - NFEAT)];
    lw[i] = (__bf16)w;
  }
  __syncthreads();

  const int lane = tid & 31;
  const int pair = blockIdx.x * 8 + (tid >> 5);
  const int npairs = (nnodes + 31) >> 5;
  if (pair >= npairs) return;
  const int node_base = pair * 32;
  const int arow0 = min(node_base +      (lane & 15), nnodes - 1);
  const int arow1 = min(node_base + 16 + (lane & 15), nnodes - 1);
  const int khalf = (lane >> 4) * 8;

  v8f acc0[8] = {};
  v8f acc1[8] = {};
#pragma unroll
  for (int ks = 0; ks < 8; ++ks) {
    const int ko = (ks < 4) ? ks * 32 : (ks - 4) * 32;
    const float* base = (ks < 4) ? agg : x;
    const v16bf a0 = frag_from_f32(base + (size_t)arow0 * NFEAT + ko + khalf);
    const v16bf a1 = frag_from_f32(base + (size_t)arow1 * NFEAT + ko + khalf);
#pragma unroll
    for (int t = 0; t < 8; ++t) {
      const v16bf b =
          frag_from_lds(&lw[(t * 16 + (lane & 15)) * 256 + ks * 32 + khalf]);
      acc0[t] = __builtin_amdgcn_wmma_f32_16x16x32_bf16(
          false, a0, false, b, (short)0, acc0[t], false, false);
      acc1[t] = __builtin_amdgcn_wmma_f32_16x16x32_bf16(
          false, a1, false, b, (short)0, acc1[t], false, false);
    }
  }

  // Epilogue: bias, L1-normalize per row (half-wave shfl reduction), relu.
  // C layout: lane<16 -> (M=i, N=lane); lane>=16 -> (M=8+i, N=lane-16).
  const int ncol = lane & 15;
  const int mofs = (lane >> 4) * 8;
  float ps0[8], ps1[8];
#pragma unroll
  for (int i = 0; i < 8; ++i) { ps0[i] = 0.f; ps1[i] = 0.f; }
#pragma unroll
  for (int t = 0; t < 8; ++t) {
    const float bias = bl[t * 16 + ncol] + br[t * 16 + ncol];
#pragma unroll
    for (int i = 0; i < 8; ++i) {
      acc0[t][i] += bias; ps0[i] += fabsf(acc0[t][i]);
      acc1[t][i] += bias; ps1[i] += fabsf(acc1[t][i]);
    }
  }
#pragma unroll
  for (int off = 1; off < 16; off <<= 1)
#pragma unroll
    for (int i = 0; i < 8; ++i) {
      ps0[i] += __shfl_xor(ps0[i], off, 32);
      ps1[i] += __shfl_xor(ps1[i], off, 32);
    }

#pragma unroll
  for (int m2 = 0; m2 < 2; ++m2) {
    const v8f*   acc = m2 ? acc1 : acc0;
    const float* ps  = m2 ? ps1  : ps0;
    const int    nb  = node_base + m2 * 16;
#pragma unroll
    for (int i = 0; i < 8; ++i) {
      const int node = nb + mofs + i;
      if (node >= nnodes) continue;
      const float inv = 1.0f / fmaxf(ps[i], 1e-12f);
#pragma unroll
      for (int t = 0; t < 8; ++t)
        h[(size_t)node * NHID + t * 16 + ncol] = fmaxf(acc[t][i] * inv, 0.f);
    }
  }
}

// ---------------------------------------------------------------------------
// Layer 2: out = log_softmax([agg|h](32x256) @ [Wl2^T;Wr2^T](256x64) + b)
// Same dual-M-tile scheme; 8 K-steps x 4 N-tiles x 2 = 64 WMMA per wave.
// ---------------------------------------------------------------------------
__global__ __launch_bounds__(256) void sage_gemm2(
    const float* __restrict__ agg, const float* __restrict__ h,
    const float* __restrict__ Wl, const float* __restrict__ bl,
    const float* __restrict__ Wr, const float* __restrict__ br,
    float* __restrict__ out, int nnodes)
{
  __shared__ __align__(16) __bf16 lw[NCLASS * 256];  // 32 KB
  const int tid = threadIdx.x;
#pragma unroll 2
  for (int i = tid; i < NCLASS * 256; i += 256) {
    const int n = i >> 8, k = i & 255;
    const float w = (k < NHID) ? Wl[n * NHID + k] : Wr[n * NHID + (k - NHID)];
    lw[i] = (__bf16)w;
  }
  __syncthreads();

  const int lane = tid & 31;
  const int pair = blockIdx.x * 8 + (tid >> 5);
  const int npairs = (nnodes + 31) >> 5;
  if (pair >= npairs) return;
  const int node_base = pair * 32;
  const int arow0 = min(node_base +      (lane & 15), nnodes - 1);
  const int arow1 = min(node_base + 16 + (lane & 15), nnodes - 1);
  const int khalf = (lane >> 4) * 8;

  v8f acc0[4] = {};
  v8f acc1[4] = {};
#pragma unroll
  for (int ks = 0; ks < 8; ++ks) {
    const int ko = (ks < 4) ? ks * 32 : (ks - 4) * 32;
    const float* base = (ks < 4) ? agg : h;
    const v16bf a0 = frag_from_f32(base + (size_t)arow0 * NHID + ko + khalf);
    const v16bf a1 = frag_from_f32(base + (size_t)arow1 * NHID + ko + khalf);
#pragma unroll
    for (int t = 0; t < 4; ++t) {
      const v16bf b =
          frag_from_lds(&lw[(t * 16 + (lane & 15)) * 256 + ks * 32 + khalf]);
      acc0[t] = __builtin_amdgcn_wmma_f32_16x16x32_bf16(
          false, a0, false, b, (short)0, acc0[t], false, false);
      acc1[t] = __builtin_amdgcn_wmma_f32_16x16x32_bf16(
          false, a1, false, b, (short)0, acc1[t], false, false);
    }
  }

  const int ncol = lane & 15;
  const int mofs = (lane >> 4) * 8;
  float pm0[8], pm1[8];
#pragma unroll
  for (int i = 0; i < 8; ++i) { pm0[i] = -3.402823466e+38f; pm1[i] = -3.402823466e+38f; }
#pragma unroll
  for (int t = 0; t < 4; ++t) {
    const float bias = bl[t * 16 + ncol] + br[t * 16 + ncol];
#pragma unroll
    for (int i = 0; i < 8; ++i) {
      acc0[t][i] += bias; pm0[i] = fmaxf(pm0[i], acc0[t][i]);
      acc1[t][i] += bias; pm1[i] = fmaxf(pm1[i], acc1[t][i]);
    }
  }
#pragma unroll
  for (int off = 1; off < 16; off <<= 1)
#pragma unroll
    for (int i = 0; i < 8; ++i) {
      pm0[i] = fmaxf(pm0[i], __shfl_xor(pm0[i], off, 32));
      pm1[i] = fmaxf(pm1[i], __shfl_xor(pm1[i], off, 32));
    }

  float pe0[8], pe1[8];
#pragma unroll
  for (int i = 0; i < 8; ++i) { pe0[i] = 0.f; pe1[i] = 0.f; }
#pragma unroll
  for (int t = 0; t < 4; ++t)
#pragma unroll
    for (int i = 0; i < 8; ++i) {
      pe0[i] += __expf(acc0[t][i] - pm0[i]);
      pe1[i] += __expf(acc1[t][i] - pm1[i]);
    }
#pragma unroll
  for (int off = 1; off < 16; off <<= 1)
#pragma unroll
    for (int i = 0; i < 8; ++i) {
      pe0[i] += __shfl_xor(pe0[i], off, 32);
      pe1[i] += __shfl_xor(pe1[i], off, 32);
    }

#pragma unroll
  for (int m2 = 0; m2 < 2; ++m2) {
    const v8f*   acc = m2 ? acc1 : acc0;
    const float* pm  = m2 ? pm1  : pm0;
    const float* pe  = m2 ? pe1  : pe0;
    const int    nb  = node_base + m2 * 16;
#pragma unroll
    for (int i = 0; i < 8; ++i) {
      const int node = nb + mofs + i;
      if (node >= nnodes) continue;
      const float lse = pm[i] + __logf(pe[i]);
#pragma unroll
      for (int t = 0; t < 4; ++t)
        out[(size_t)node * NCLASS + t * 16 + ncol] = acc[t][i] - lse;
    }
  }
}

// ---------------------------------------------------------------------------
extern "C" void kernel_launch(void* const* d_in, const int* in_sizes, int n_in,
                              void* d_out, int out_size, void* d_ws, size_t ws_size,
                              hipStream_t stream) {
  const float* x    = (const float*)d_in[0];
  const int*   rows = (const int*)d_in[1];
  const int*   cols = (const int*)d_in[2];
  const float* vals = (const float*)d_in[3];
  const float* Wl1  = (const float*)d_in[4];
  const float* bl1  = (const float*)d_in[5];
  const float* Wr1  = (const float*)d_in[6];
  const float* br1  = (const float*)d_in[7];
  const float* Wl2  = (const float*)d_in[8];
  const float* bl2  = (const float*)d_in[9];
  const float* Wr2  = (const float*)d_in[10];
  const float* br2  = (const float*)d_in[11];

  const int nnodes = in_sizes[0] / NFEAT;   // 100000
  const int nedges = in_sizes[1];           // 1600000

  float* agg = (float*)d_ws;                          // N*128 f32 (reused both layers)
  float* h   = agg + (size_t)nnodes * NHID;           // N*128 f32

  const long nz = (long)nnodes * NHID;
  const int zblocks = (int)((nz / 4 + 255) / 256);
  const int sblocks = (int)(((nedges + 127) / 128 + 7) / 8);
  const int gblocks = (((nnodes + 31) / 32) + 7) / 8;

  // Layer 1
  sage_zero <<<zblocks, 256, 0, stream>>>(agg, nz);
  sage_spmm <<<sblocks, 256, 0, stream>>>(rows, cols, vals, x, agg, nedges);
  sage_gemm1<<<gblocks, 256, 0, stream>>>(agg, x, Wl1, bl1, Wr1, br1, h, nnodes);
  // Layer 2
  sage_zero <<<zblocks, 256, 0, stream>>>(agg, nz);
  sage_spmm <<<sblocks, 256, 0, stream>>>(rows, cols, vals, h, agg, nedges);
  sage_gemm2<<<gblocks, 256, 0, stream>>>(agg, h, Wl2, bl2, Wr2, br2,
                                          (float*)d_out, nnodes);
}
